// Attentional_Factorization_Machine_68917045232364
// MI455X (gfx1250) — matrix operand order repaired and checked by reading, not compiled, not run on hardware
//
#include <hip/hip_runtime.h>
#include <hip/hip_bf16.h>

typedef __attribute__((ext_vector_type(16))) _Float16 v16h;
typedef __attribute__((ext_vector_type(8)))  _Float16 v8h;
typedef __attribute__((ext_vector_type(4)))  _Float16 v4h;
typedef __attribute__((ext_vector_type(8)))  float    v8f;
typedef __attribute__((ext_vector_type(2)))  float    v2f;

#define NF    32      // fields
#define NB    2048    // batch
#define NK    64      // embedding dim
#define NH    64      // attention hidden
#define NPAIR 496     // NF*(NF-1)/2

// LDS: V[f][b'][k] in f16, k-blocks of 8 halves XOR-swizzled by (b'&7)
// row stride = 64 halves (8 blocks of 16B) -> 32*16*64*2 = 64KB
__global__ __launch_bounds__(128) void afm_main_kernel(
    const int*   __restrict__ x,      // [NF, NB]
    const float* __restrict__ emb_v,  // [VOCAB, NK]
    const float* __restrict__ at_w,   // [NK, NH]
    const float* __restrict__ at_b,   // [NH]
    const float* __restrict__ at_h,   // [NH]
    float*       __restrict__ pooled) // [NB, NK] accumulated via atomics
{
    __shared__ _Float16 Vs[NF * 16 * 64];   // 65536 bytes

    const int tid   = threadIdx.x;
    const int tile  = blockIdx.x >> 2;      // 0..127 batch tile
    const int chunk = blockIdx.x & 3;       // 0..3 pair chunk

    // ---- gather V rows for this batch tile into LDS (f32 -> f16) ----
    for (int idx = tid; idx < NF * 16 * 16; idx += 128) {
        int row = idx >> 4;                 // f*16 + b'
        int c4  = idx & 15;                 // float4 column (4 halves)
        int f   = row >> 4;
        int bp  = row & 15;
        int b   = tile * 16 + bp;
        long e  = (long)x[f * NB + b];
        float4 src = reinterpret_cast<const float4*>(emb_v + e * NK)[c4];
        int kb  = c4 >> 1;                  // 16B block index 0..7
        int kbs = kb ^ (bp & 7);            // swizzle
        _Float16* dst = &Vs[row * 64 + kbs * 8 + (c4 & 1) * 4];
        v4h d; d[0] = (_Float16)src.x; d[1] = (_Float16)src.y;
               d[2] = (_Float16)src.z; d[3] = (_Float16)src.w;
        *reinterpret_cast<v4h*>(dst) = d;
    }

    const int lane = tid & 31;
    const int wv   = tid >> 5;
    const int g    = lane >> 4;   // half-wave
    const int n    = lane & 15;   // column (B/C/D) and batch row (A)

    // ---- preload at_w B-tiles in WMMA B layout: elem e <-> K = 32t+8g+(e<8?e:e+8), col = 16ht+n ----
    v16h bt[2][4];
    v8f  cbias[4];                // bias splat as WMMA C operand (free bias add)
    float ath[4];
    for (int t = 0; t < 2; ++t)
        for (int ht = 0; ht < 4; ++ht)
            for (int e = 0; e < 16; ++e) {
                int k = 32 * t + 8 * g + (e < 8 ? e : e + 8);
                bt[t][ht][e] = (_Float16)at_w[k * NH + 16 * ht + n];
            }
    for (int ht = 0; ht < 4; ++ht) {
        float bv = at_b[16 * ht + n];
        for (int r = 0; r < 8; ++r) cbias[ht][r] = bv;
        ath[ht] = at_h[16 * ht + n];
    }

    __syncthreads();

    float pool[2][16];
    for (int t = 0; t < 2; ++t)
        for (int e = 0; e < 16; ++e) pool[t][e] = 0.f;

    // pair range for this wave: q0 .. q0+30
    int i = 0, j;
    {
        int qq = (chunk * 4 + wv) * 31;
        int rem = NF - 1;
        while (qq >= rem) { qq -= rem; ++i; --rem; }
        j = i + 1 + qq;
    }

    union au { v16h v; v8h h[2]; unsigned u[8]; };
    union sv { float f[8]; v2f v2[4]; double d[4]; };

    for (int pr = 0; pr < 31; ++pr) {
        // ---- build A tiles: VV = V_i * V_j in 16-bit A layout ----
        const _Float16* Ri = &Vs[(i * 16 + n) * 64];
        const _Float16* Rj = &Vs[(j * 16 + n) * 64];
        const int sw = (n & 7);
        au a[2];
        #pragma unroll
        for (int t = 0; t < 2; ++t) {
            int kb0 = (4 * t + g) ^ sw;          // k = 32t + 8g
            int kb1 = (4 * t + 2 + g) ^ sw;      // k = 32t + 16 + 8g
            v8h vi0 = *reinterpret_cast<const v8h*>(Ri + kb0 * 8);
            v8h vj0 = *reinterpret_cast<const v8h*>(Rj + kb0 * 8);
            v8h vi1 = *reinterpret_cast<const v8h*>(Ri + kb1 * 8);
            v8h vj1 = *reinterpret_cast<const v8h*>(Rj + kb1 * 8);
            a[t].h[0] = vi0 * vj0;               // v_pk_mul_f16
            a[t].h[1] = vi1 * vj1;
        }

        // ---- hid = relu(VV @ at_w + b); s partial = hid . at_h ----
        sv s;
        #pragma unroll
        for (int r = 0; r < 8; ++r) s.f[r] = 0.f;
        #pragma unroll
        for (int ht = 0; ht < 4; ++ht) {
            v8f c = __builtin_amdgcn_wmma_f32_16x16x32_f16(false, a[0].v, false, bt[0][ht],
                                                           (short)0, cbias[ht], false, false);
            c = __builtin_amdgcn_wmma_f32_16x16x32_f16(false, a[1].v, false, bt[1][ht],
                                                       (short)0, c, false, false);
            #pragma unroll
            for (int r = 0; r < 8; ++r) {
                // single-instruction ReLU: med3(x, 0, +MAX) -> v_med3_num_f32
                float hid = __builtin_amdgcn_fmed3f(c[r], 0.f, 3.402823466e+38f);
                s.f[r] = fmaf(hid, ath[ht], s.f[r]);
            }
        }
        // butterfly over the 16 lanes sharing g: 64-bit shuffles + packed adds
        #pragma unroll
        for (int m = 1; m < 16; m <<= 1)
            #pragma unroll
            for (int q = 0; q < 4; ++q) {
                double o = __shfl_xor(s.d[q], m, 32);
                s.v2[q] += __builtin_bit_cast(v2f, o);
            }
        // fetch score for batch row n (may live in the other half-wave)
        float own = s.f[n & 7];
        float oth = __shfl_xor(own, 16, 32);
        float sc  = ((n >> 3) == g) ? own : oth;

        // ---- pooled += VV * score via v_fma_mix_f32 (f16 operand read in place) ----
        #pragma unroll
        for (int t = 0; t < 2; ++t)
            #pragma unroll
            for (int d = 0; d < 8; ++d) {
                asm("v_fma_mix_f32 %0, %1, %2, %0 op_sel:[0,0,0] op_sel_hi:[1,0,0]"
                    : "+v"(pool[t][2 * d]) : "v"(a[t].u[d]), "v"(sc));
                asm("v_fma_mix_f32 %0, %1, %2, %0 op_sel:[1,0,0] op_sel_hi:[1,0,0]"
                    : "+v"(pool[t][2 * d + 1]) : "v"(a[t].u[d]), "v"(sc));
            }

        ++j;
        if (j == NF) { ++i; j = i + 1; }
    }

    // ---- flush pooled partials ----
    {
        int b = tile * 16 + n;
        for (int t = 0; t < 2; ++t)
            for (int e = 0; e < 16; ++e) {
                int k = 32 * t + 8 * g + (e < 8 ? e : e + 8);
                atomicAdd(&pooled[b * NK + k], pool[t][e]);
            }
    }
}

__global__ __launch_bounds__(256) void afm_final_kernel(
    const int*   __restrict__ x,
    const float* __restrict__ pooled,
    const float* __restrict__ p,
    const float* __restrict__ w0,
    const float* __restrict__ w1,
    float*       __restrict__ out)
{
    int b = blockIdx.x * blockDim.x + threadIdx.x;
    if (b >= NB) return;
    float acc = w0[0];
    for (int k = 0; k < NK; ++k)
        acc = fmaf(pooled[b * NK + k], p[k], acc);
    for (int f = 0; f < NF; ++f)
        acc += w1[x[f * NB + b]];
    out[b] = 1.f / (1.f + expf(-acc));
}

extern "C" void kernel_launch(void* const* d_in, const int* in_sizes, int n_in,
                              void* d_out, int out_size, void* d_ws, size_t ws_size,
                              hipStream_t stream) {
    const int*   x     = (const int*)  d_in[0];
    const float* emb_v = (const float*)d_in[1];
    const float* at_w  = (const float*)d_in[2];
    const float* at_b  = (const float*)d_in[3];
    const float* at_h  = (const float*)d_in[4];
    const float* p     = (const float*)d_in[5];
    const float* w0    = (const float*)d_in[6];
    const float* w1    = (const float*)d_in[7];
    float* pooled = (float*)d_ws;                    // NB*NK floats = 512KB
    float* out    = (float*)d_out;

    hipMemsetAsync(pooled, 0, (size_t)NB * NK * sizeof(float), stream);
    afm_main_kernel<<<dim3(512), dim3(128), 0, stream>>>(x, emb_v, at_w, at_b, at_h, pooled);
    afm_final_kernel<<<dim3(8), dim3(256), 0, stream>>>(x, pooled, p, w0, w1, out);
}
